// MultiHeadCrossAttention_4131758538855
// MI455X (gfx1250) — matrix operand order repaired and checked by reading, not compiled, not run on hardware
//
#include <hip/hip_runtime.h>
#include <hip/hip_bf16.h>
#include <math.h>

typedef __attribute__((ext_vector_type(16))) _Float16 v16h;
typedef __attribute__((ext_vector_type(8)))  float    v8f;
typedef __attribute__((ext_vector_type(4)))  _Float16 v4h;

union Frag16 { v16h h; uint4 q[2]; };

#define DIMC   1024
#define HEADS  16
#define HDIM   64
#define NX     1024
#define NKV    2048
#define SCALE_F 0.125f   // HEAD_DIM^-0.5 = 1/8, folded into Q

// Async global->LDS copy of 16B per lane (CDNA5, tracked by ASYNCcnt).
// ldsOff = byte offset within LDS space (low 32 bits of generic pointer).
__device__ __forceinline__ void async_b128(unsigned ldsOff, const void* gptr) {
  asm volatile("global_load_async_to_lds_b128 %0, %1, off"
               :: "v"(ldsOff), "v"((unsigned long long)gptr) : "memory");
}
__device__ __forceinline__ unsigned lds_off(const void* p) {
  return (unsigned)(unsigned long long)p;   // addr[31:0] = LDS offset (ISA aperture rule)
}
__device__ __forceinline__ void wait_async_le4() {
  asm volatile("s_wait_asynccnt 4" ::: "memory");
}
__device__ __forceinline__ void wait_async_le2() {
  asm volatile("s_wait_asynccnt 2" ::: "memory");
}
__device__ __forceinline__ void wait_async_0() {
  asm volatile("s_wait_asynccnt 0" ::: "memory");
}

// -------------------------------------------------------------------------
// Build f16 kv-source (concat of x and complement along sequence) and f16 x.
// -------------------------------------------------------------------------
__global__ __launch_bounds__(256) void fill_inputs(
    const float* __restrict__ x, const float* __restrict__ comp,
    _Float16* __restrict__ kvsrcH, _Float16* __restrict__ xH)
{
  int id = blockIdx.x * 256 + threadIdx.x;      // 8192*256 ids, 4 floats each
  int m  = id >> 8;                             // row: b*2048 + j
  int c  = (id & 255) * 4;
  int b  = m >> 11;
  int j  = m & 2047;
  const float* src = (j < 1024)
      ? (x    + ((size_t)(b * 1024 + j))          * DIMC + c)
      : (comp + ((size_t)(b * 1024 + (j - 1024))) * DIMC + c);
  float4 v = *(const float4*)src;
  v4h hv = { (_Float16)v.x, (_Float16)v.y, (_Float16)v.z, (_Float16)v.w };
  *(v4h*)(kvsrcH + (size_t)m * DIMC + c) = hv;
  if (j < 1024)
    *(v4h*)(xH + ((size_t)(b * 1024 + j)) * DIMC + c) = hv;
}

// -------------------------------------------------------------------------
// Transpose-convert weight: W[K][N] f32 -> Wt[N][K] f16 (coalesced both sides)
// -------------------------------------------------------------------------
__global__ __launch_bounds__(256) void transpose_w(
    const float* __restrict__ W, _Float16* __restrict__ Wt, int K, int N)
{
  __shared__ float tile[32][33];
  int n0 = blockIdx.x * 32, k0 = blockIdx.y * 32;
  int tx = threadIdx.x, ty = threadIdx.y;
  #pragma unroll
  for (int i = 0; i < 32; i += 8)
    tile[ty + i][tx] = W[(size_t)(k0 + ty + i) * N + (n0 + tx)];
  __syncthreads();
  #pragma unroll
  for (int i = 0; i < 32; i += 8)
    Wt[(size_t)(n0 + ty + i) * K + (k0 + tx)] = (_Float16)tile[tx][ty + i];
}

// -------------------------------------------------------------------------
// WMMA GEMM: C[M][N] = A[M][K] * Bt[N][K]^T   (f16 in, f32 accumulate)
// Block tile 128x128, K-tile 32, double-buffered LDS filled by async
// global->LDS DMA (ASYNCcnt). 8 waves in 2x4 grid, 64x32 per wave.
// mode 0: outH = f16(C * scale)                        (Q projection)
// mode 1: split KV: cols<1024 -> kh[b][j][c]; cols>=1024 -> vt[b][h][d][j]
// mode 2: outF = C + bias                              (output projection)
// -------------------------------------------------------------------------
#define SA 40   // LDS row stride in halves (32 + 8 pad -> 80B, conflict-free)

__global__ __launch_bounds__(256) void gemm_wmma(
    const _Float16* __restrict__ A, const _Float16* __restrict__ Bt,
    int M, int N, int K, int mode, float scale, const float* __restrict__ bias,
    _Float16* __restrict__ outH, _Float16* __restrict__ khO,
    _Float16* __restrict__ vtO, float* __restrict__ outF)
{
  __shared__ _Float16 ldsA[2][128 * SA];
  __shared__ _Float16 ldsB[2][128 * SA];

  const int tid  = threadIdx.x;
  const int wave = tid >> 5, lane = tid & 31;
  const int lh   = lane & 15, hi = lane >> 4;
  const int mBase = blockIdx.y * 128, nBase = blockIdx.x * 128;
  const int waveM = (wave >> 2) * 64, waveN = (wave & 3) * 32;

  v8f zero = {};
  v8f acc[4][2];
  #pragma unroll
  for (int mi = 0; mi < 4; ++mi)
    #pragma unroll
    for (int ni = 0; ni < 2; ++ni) acc[mi][ni] = zero;

  const int rA = tid >> 1;            // 128 rows x 2 half-row chunks
  const int cA = (tid & 1) * 16;
  const _Float16* gA = A  + (size_t)(mBase + rA) * K + cA;
  const _Float16* gB = Bt + (size_t)(nBase + rA) * K + cA;
  unsigned sAoff[2], sBoff[2];
  sAoff[0] = lds_off(&ldsA[0][rA * SA + cA]);
  sAoff[1] = lds_off(&ldsA[1][rA * SA + cA]);
  sBoff[0] = lds_off(&ldsB[0][rA * SA + cA]);
  sBoff[1] = lds_off(&ldsB[1][rA * SA + cA]);

  const int ntiles = K >> 5;
  // prologue: prefetch tile 0 (4 async ops per wave)
  async_b128(sAoff[0],      gA);
  async_b128(sAoff[0] + 16, gA + 8);
  async_b128(sBoff[0],      gB);
  async_b128(sBoff[0] + 16, gB + 8);

  for (int kt = 0; kt < ntiles; ++kt) {
    const int p = kt & 1;
    if (kt + 1 < ntiles) {
      const _Float16* nA = gA + (size_t)(kt + 1) * 32;
      const _Float16* nB = gB + (size_t)(kt + 1) * 32;
      async_b128(sAoff[p ^ 1],      nA);
      async_b128(sAoff[p ^ 1] + 16, nA + 8);
      async_b128(sBoff[p ^ 1],      nB);
      async_b128(sBoff[p ^ 1] + 16, nB + 8);
      wait_async_le4();           // tile kt resident, prefetch still in flight
    } else {
      wait_async_0();
    }
    __syncthreads();              // all waves' tile-kt data visible

    Frag16 bf[2];
    #pragma unroll
    for (int ni = 0; ni < 2; ++ni) {
      const _Float16* bp = &ldsB[p][(waveN + ni * 16 + lh) * SA + hi * 8];
      bf[ni].q[0] = *(const uint4*)(bp);
      bf[ni].q[1] = *(const uint4*)(bp + 16);
    }
    #pragma unroll
    for (int mi = 0; mi < 4; ++mi) {
      Frag16 af;
      const _Float16* ap = &ldsA[p][(waveM + mi * 16 + lh) * SA + hi * 8];
      af.q[0] = *(const uint4*)(ap);
      af.q[1] = *(const uint4*)(ap + 16);
      #pragma unroll
      for (int ni = 0; ni < 2; ++ni)
        acc[mi][ni] = __builtin_amdgcn_wmma_f32_16x16x32_f16(
            false, af.h, false, bf[ni].h, (short)0, acc[mi][ni], false, false);
    }
    __syncthreads();              // reads done before buffer p is refilled
  }

  const int mW = mBase + waveM, nW = nBase + waveN;
  #pragma unroll
  for (int mi = 0; mi < 4; ++mi)
    #pragma unroll
    for (int ni = 0; ni < 2; ++ni)
      #pragma unroll
      for (int j = 0; j < 8; ++j) {
        int row = mW + mi * 16 + j + 8 * hi;       // C layout: vgpr j -> M=j(+8)
        int col = nW + ni * 16 + lh;               // lane%16 -> N
        float v = acc[mi][ni][j];
        if (mode == 0) {
          outH[(size_t)row * N + col] = (_Float16)(v * scale);
        } else if (mode == 1) {
          int b = row >> 11, jj = row & 2047;
          if (col < 1024) {
            khO[((size_t)(b * NKV + jj)) * DIMC + col] = (_Float16)v;
          } else {
            int c2 = col - 1024;
            vtO[((size_t)((b * HEADS + (c2 >> 6)) * HDIM + (c2 & 63))) * NKV + jj]
                = (_Float16)v;
          }
        } else {
          outF[(size_t)row * N + col] = v + bias[col];
        }
      }
}

// -------------------------------------------------------------------------
// Flash attention: block = (b, h, 128 q rows). Each wave owns 16 full rows.
// K/Vt tiles double-buffered in LDS via async DMA. S = Q Kt (WMMA), online
// softmax in registers, P relayout via per-wave LDS, O += P V (WMMA).
// Epilogue: O/rowsum + residual x -> f16.
// -------------------------------------------------------------------------
__global__ __launch_bounds__(256) void attn_wmma(
    const _Float16* __restrict__ qH, const _Float16* __restrict__ kH,
    const _Float16* __restrict__ vtH, const float* __restrict__ x,
    _Float16* __restrict__ aoH)
{
  __shared__ _Float16 ldsK[2][64 * 72];   // [kj][d]   64x64, stride 72
  __shared__ _Float16 ldsV[2][64 * 72];   // [d][kj]   64x64, stride 72
  __shared__ _Float16 ldsP[8 * 16 * 72];  // per-wave 16x64 P slice

  const int tid  = threadIdx.x;
  const int wave = tid >> 5, lane = tid & 31;
  const int lh   = lane & 15, hi = lane >> 4;

  const int bid = blockIdx.x;             // 512 blocks: b*128 + h*8 + qt
  const int qt  = bid & 7;
  const int h   = (bid >> 3) & 15;
  const int b   = bid >> 7;

  const int rT = tid >> 2;
  const int cT = (tid & 3) * 16;
  const _Float16* gK = kH  + ((size_t)(b * NKV + rT)) * DIMC + h * HDIM + cT;
  const _Float16* gV = vtH + ((size_t)((b * HEADS + h) * HDIM + rT)) * NKV + cT;
  unsigned kOff[2], vOff[2];
  kOff[0] = lds_off(&ldsK[0][rT * 72 + cT]);
  kOff[1] = lds_off(&ldsK[1][rT * 72 + cT]);
  vOff[0] = lds_off(&ldsV[0][rT * 72 + cT]);
  vOff[1] = lds_off(&ldsV[1][rT * 72 + cT]);

  // prefetch tile 0 (2 async ops per wave)
  async_b128(kOff[0], gK);
  async_b128(vOff[0], gV);

  // Q fragments loaded once, straight into A layout (SCALE already folded in)
  Frag16 qf[2];
  {
    const _Float16* qrow =
        qH + ((size_t)(b * NX + qt * 128 + wave * 16 + lh)) * DIMC + h * HDIM;
    #pragma unroll
    for (int kc = 0; kc < 2; ++kc) {
      qf[kc].q[0] = *(const uint4*)(qrow + kc * 32 + hi * 8);
      qf[kc].q[1] = *(const uint4*)(qrow + kc * 32 + 16 + hi * 8);
    }
  }

  v8f zero = {};
  v8f o[4];
  #pragma unroll
  for (int di = 0; di < 4; ++di) o[di] = zero;
  float rm[8], rs[8];
  #pragma unroll
  for (int j = 0; j < 8; ++j) { rm[j] = -1e30f; rs[j] = 0.0f; }

  _Float16* pw = ldsP + wave * (16 * 72);
  const int ntiles = NKV / 64;

  for (int kt = 0; kt < ntiles; ++kt) {
    const int p = kt & 1;
    if (kt + 1 < ntiles) {
      const int kj1 = (kt + 1) * 64;
      async_b128(kOff[p ^ 1], gK + (size_t)kj1 * DIMC);
      async_b128(vOff[p ^ 1], gV + kj1);
      wait_async_le2();
    } else {
      wait_async_0();
    }
    __syncthreads();

    // ---- S = Q * K^T for this 16x64 stripe -------------------------------
    v8f s[4];
    #pragma unroll
    for (int ni = 0; ni < 4; ++ni) s[ni] = zero;
    #pragma unroll
    for (int kc = 0; kc < 2; ++kc)
      #pragma unroll
      for (int ni = 0; ni < 4; ++ni) {
        Frag16 bf;
        const _Float16* bp = &ldsK[p][(ni * 16 + lh) * 72 + kc * 32 + hi * 8];
        bf.q[0] = *(const uint4*)(bp);
        bf.q[1] = *(const uint4*)(bp + 16);
        s[ni] = __builtin_amdgcn_wmma_f32_16x16x32_f16(
            false, qf[kc].h, false, bf.h, (short)0, s[ni], false, false);
      }

    // ---- online softmax (row reductions stay inside a 16-lane half) ------
    #pragma unroll
    for (int j = 0; j < 8; ++j) {
      float mj = fmaxf(fmaxf(s[0][j], s[1][j]), fmaxf(s[2][j], s[3][j]));
      mj = fmaxf(mj, __shfl_xor(mj, 1));
      mj = fmaxf(mj, __shfl_xor(mj, 2));
      mj = fmaxf(mj, __shfl_xor(mj, 4));
      mj = fmaxf(mj, __shfl_xor(mj, 8));
      float mn = fmaxf(rm[j], mj);
      float alpha = __expf(rm[j] - mn);
      rm[j] = mn;
      float sum = 0.0f;
      #pragma unroll
      for (int ni = 0; ni < 4; ++ni) {
        float pv = __expf(s[ni][j] - mn);
        s[ni][j] = pv;
        sum += pv;
      }
      sum += __shfl_xor(sum, 1);
      sum += __shfl_xor(sum, 2);
      sum += __shfl_xor(sum, 4);
      sum += __shfl_xor(sum, 8);
      rs[j] = rs[j] * alpha + sum;
      #pragma unroll
      for (int di = 0; di < 4; ++di) o[di][j] *= alpha;
    }

    // ---- P: C layout -> A layout via per-wave LDS slice ------------------
    #pragma unroll
    for (int ni = 0; ni < 4; ++ni)
      #pragma unroll
      for (int j = 0; j < 8; ++j)
        pw[(j + 8 * hi) * 72 + ni * 16 + lh] = (_Float16)s[ni][j];
    // same-wave LDS RAW: DS ops in order per wave; compiler waits DScnt

    // ---- O += P * V ------------------------------------------------------
    #pragma unroll
    for (int kc = 0; kc < 2; ++kc) {
      Frag16 af;
      const _Float16* ap = pw + lh * 72 + kc * 32;
      af.q[0] = *(const uint4*)(ap + hi * 8);
      af.q[1] = *(const uint4*)(ap + 16 + hi * 8);
      #pragma unroll
      for (int di = 0; di < 4; ++di) {
        Frag16 bf;
        const _Float16* bp = &ldsV[p][(di * 16 + lh) * 72 + kc * 32 + hi * 8];
        bf.q[0] = *(const uint4*)(bp);
        bf.q[1] = *(const uint4*)(bp + 16);
        o[di] = __builtin_amdgcn_wmma_f32_16x16x32_f16(
            false, af.h, false, bf.h, (short)0, o[di], false, false);
      }
    }
    __syncthreads();              // reads done before buffer p is refilled
  }

  // ---- normalize, add residual, store f16 --------------------------------
  #pragma unroll
  for (int j = 0; j < 8; ++j) {
    float inv = 1.0f / rs[j];
    int qg = b * NX + qt * 128 + wave * 16 + j + 8 * hi;
    #pragma unroll
    for (int di = 0; di < 4; ++di) {
      int c = h * HDIM + di * 16 + lh;
      float val = o[di][j] * inv + x[(size_t)qg * DIMC + c];
      aoH[(size_t)qg * DIMC + c] = (_Float16)val;
    }
  }
}

// -------------------------------------------------------------------------
extern "C" void kernel_launch(void* const* d_in, const int* in_sizes, int n_in,
                              void* d_out, int out_size, void* d_ws, size_t ws_size,
                              hipStream_t stream) {
  const float* x     = (const float*)d_in[0];
  const float* comp  = (const float*)d_in[1];
  const float* Wq    = (const float*)d_in[2];
  const float* Wkv   = (const float*)d_in[3];
  const float* Wproj = (const float*)d_in[4];
  const float* bproj = (const float*)d_in[5];
  float* out = (float*)d_out;

  // workspace carve (f16 elements; ~80 MB total)
  _Float16* p = (_Float16*)d_ws;
  _Float16* qH     = p; p += (size_t)4096 * 1024;   // [B*Nx][DIM]  Q * SCALE
  _Float16* kvsrcH = p; p += (size_t)8192 * 1024;   // [B*Nkv][DIM]
  _Float16* xH     = p; p += (size_t)4096 * 1024;   // [B*Nx][DIM]
  _Float16* kH     = p; p += (size_t)8192 * 1024;   // [B][Nkv][H*D]
  _Float16* vtH    = p; p += (size_t)8192 * 1024;   // [B][H][D][Nkv]
  _Float16* aoH    = p; p += (size_t)4096 * 1024;   // attn out + residual
  _Float16* WqT    = p; p += (size_t)1024 * 1024;   // [N][K]
  _Float16* WkvT   = p; p += (size_t)2048 * 1024;   // [2N][K]
  _Float16* WprojT = p; p += (size_t)1024 * 1024;   // [N][K]

  fill_inputs<<<8192, 256, 0, stream>>>(x, comp, kvsrcH, xH);
  transpose_w<<<dim3(32, 32), dim3(32, 8), 0, stream>>>(Wq,    WqT,    1024, 1024);
  transpose_w<<<dim3(64, 32), dim3(32, 8), 0, stream>>>(Wkv,   WkvT,   1024, 2048);
  transpose_w<<<dim3(32, 32), dim3(32, 8), 0, stream>>>(Wproj, WprojT, 1024, 1024);

  // Q = x @ Wq * SCALE
  gemm_wmma<<<dim3(8, 32), 256, 0, stream>>>(xH, WqT, 4096, 1024, 1024, 0,
      SCALE_F, nullptr, qH, nullptr, nullptr, nullptr);
  // KV = kvsrc @ Wkv  -> K dense, V transposed per (b,h)
  gemm_wmma<<<dim3(16, 64), 256, 0, stream>>>(kvsrcH, WkvT, 8192, 2048, 1024, 1,
      1.0f, nullptr, nullptr, kH, vtH, nullptr);
  // flash attention + residual
  attn_wmma<<<512, 256, 0, stream>>>(qH, kH, vtH, x, aoH);
  // out = (attn + x) @ Wproj + bproj
  gemm_wmma<<<dim3(8, 32), 256, 0, stream>>>(aoH, WprojT, 4096, 1024, 1024, 2,
      1.0f, bproj, nullptr, nullptr, nullptr, out);
}
